// LinearLowbit_8701603741968
// MI455X (gfx1250) — 8-bit matrix operand order repaired and checked by reading, compiled, not run on hardware
//
#include <hip/hip_runtime.h>

typedef __attribute__((ext_vector_type(16))) int   v16i;
typedef __attribute__((ext_vector_type(8)))  float v8f;


#define WS_XQ_OFF 256l

__global__ void ll_amax_kernel(const float* __restrict__ p, long n4,
                               unsigned int* __restrict__ slot) {
  float m = 0.f;
  long i = (long)blockIdx.x * blockDim.x + threadIdx.x;
  long stride = (long)gridDim.x * blockDim.x;
  const float4* p4 = (const float4*)p;
  for (; i < n4; i += stride) {
    float4 v = p4[i];
    m = fmaxf(m, fmaxf(fmaxf(fabsf(v.x), fabsf(v.y)),
                       fmaxf(fabsf(v.z), fabsf(v.w))));
  }
  for (int off = 16; off > 0; off >>= 1)
    m = fmaxf(m, __shfl_down(m, off, 32));
  __shared__ float sm[8];
  int lane = threadIdx.x & 31, wid = threadIdx.x >> 5;
  if (lane == 0) sm[wid] = m;
  __syncthreads();
  if (threadIdx.x == 0) {
    float b = sm[0];
    for (int k = 1; k < 8; ++k) b = fmaxf(b, sm[k]);
    atomicMax(slot, __float_as_uint(b));
  }
}

__global__ void ll_init_kernel(unsigned int* __restrict__ slots) {
  if (threadIdx.x < 2) slots[threadIdx.x] = 0u;
}

__device__ __forceinline__ unsigned int ll_q1(float val, float inv_scale) {
  float v = val * inv_scale;
  float a = fminf(fabsf(v), 6.0f);
  float step = a < 2.0f ? 0.5f : (a < 4.0f ? 1.0f : 2.0f);
  float q = __builtin_rintf(a / step) * step;
  unsigned int b = __float_as_uint(q);
  unsigned int code =
      (q == 0.0f) ? 0u : (((((b >> 23) & 0xffu) - 120u) << 3) | ((b >> 20) & 7u));
  code |= (__float_as_uint(v) >> 24) & 0x80u;
  return code;
}

__global__ void ll_quant_kernel(const float* __restrict__ p, long n4,
                                const unsigned int* __restrict__ slot,
                                unsigned int* __restrict__ q8) {
  float amax = __uint_as_float(*slot);
  float scale = amax > 0.f ? amax * (1.0f / 6.0f) : 1.0f;
  float inv = 1.0f / scale;
  long i = (long)blockIdx.x * blockDim.x + threadIdx.x;
  long stride = (long)gridDim.x * blockDim.x;
  const float4* p4 = (const float4*)p;
  for (; i < n4; i += stride) {
    float4 v = p4[i];
    unsigned int packed = ll_q1(v.x, inv) | (ll_q1(v.y, inv) << 8) |
                          (ll_q1(v.z, inv) << 16) | (ll_q1(v.w, inv) << 24);
    q8[i] = packed;
  }
}

__device__ __forceinline__ void ll_async_b128(unsigned lds_off, const void* gaddr) {
  asm volatile("global_load_async_to_lds_b128 %0, %1, off"
               :: "v"(lds_off), "v"(gaddr)
               : "memory");
}
__device__ __forceinline__ void ll_wait_async() {
  asm volatile("s_wait_asynccnt 0x0" ::: "memory");
}

#define BM 128
#define BN 128
#define BK 128

__global__ __launch_bounds__(256, 1) void ll_gemm_fp8_kernel(
    const unsigned char* __restrict__ xq,
    const unsigned char* __restrict__ wq,
    const float* __restrict__ bias,
    const unsigned int* __restrict__ slots,
    float* __restrict__ out, int M, int N, int K) {
  __shared__ unsigned char lA[2][BM * BK];
  __shared__ unsigned char lB[2][BN * BK];

  float ax = __uint_as_float(slots[0]);
  float aw = __uint_as_float(slots[1]);
  float sx = ax > 0.f ? ax * (1.0f / 6.0f) : 1.0f;
  float sw = aw > 0.f ? aw * (1.0f / 6.0f) : 1.0f;
  float sc = sx * sw;

  const int m0 = blockIdx.y * BM;
  const int n0 = blockIdx.x * BN;
  const int tid = threadIdx.x;
  const int lane = tid & 31;
  const int wave = tid >> 5;
  const int wm = wave >> 1;
  const int wn = wave & 1;
  const int lane16 = lane & 15;
  const int laneHi = lane >> 4;

  const unsigned lA0 = (unsigned)(size_t)&lA[0][0];
  const unsigned lB0 = (unsigned)(size_t)&lB[0][0];
  const int row = tid >> 1;
  const int coff = (tid & 1) * 64;

  v8f acc[2][4];
#pragma unroll
  for (int im = 0; im < 2; ++im)
#pragma unroll
    for (int in = 0; in < 4; ++in)
#pragma unroll
      for (int r = 0; r < 8; ++r) acc[im][in][r] = 0.0f;

  auto stage = [&](int buf, int k0) {
    const unsigned aBase = lA0 + (unsigned)buf * (BM * BK);
    const unsigned bBase = lB0 + (unsigned)buf * (BN * BK);
#pragma unroll
    for (int i = 0; i < 4; ++i) {
      int t = tid + 256 * i;
      int r = t >> 3, c = (t & 7) * 16;
      ll_async_b128(aBase + (unsigned)(t * 16),
                    xq + (long)(m0 + r) * K + k0 + c);
      ll_async_b128(bBase + (unsigned)(t * 16),
                    wq + (long)(n0 + r) * K + k0 + c);
    }
  };

  stage(0, 0);
  ll_wait_async();
  __syncthreads();

  int p = 0;
  for (int k0 = 0; k0 < K; k0 += BK) {
    if (k0 + BK < K) stage(p ^ 1, k0 + BK);

    const unsigned char* cA = lA[p];
    const unsigned char* cB = lB[p];

    v16i afrag[2];
#pragma unroll
    for (int im = 0; im < 2; ++im) {
      const unsigned char* abase = cA + (wm * 32 + im * 16 + lane16) * BK;
#pragma unroll
      for (int v = 0; v < 16; v += 2) {
        int kk = (v >> 2) * 32 + laneHi * 16 + (v & 3) * 4;
        int2 d = *(const int2*)(abase + kk);
        afrag[im][v] = d.x;
        afrag[im][v + 1] = d.y;
      }
    }

#pragma unroll
    for (int in = 0; in < 4; ++in) {
      const unsigned char* bbase = cB + (wn * 64 + in * 16 + lane16) * BK;
      v16i bfrag;
#pragma unroll
      for (int v = 0; v < 16; v += 4) {
        int kk = (v >> 2) * 32 + laneHi * 16;
        int4 d = *(const int4*)(bbase + kk);
        bfrag[v] = d.x;
        bfrag[v + 1] = d.y;
        bfrag[v + 2] = d.z;
        bfrag[v + 3] = d.w;
      }
      acc[0][in] = __builtin_amdgcn_wmma_f32_16x16x128_fp8_fp8(
          afrag[0], bfrag, (short)0, acc[0][in], false, false);
      acc[1][in] = __builtin_amdgcn_wmma_f32_16x16x128_fp8_fp8(
          afrag[1], bfrag, (short)0, acc[1][in], false, false);
    }

    ll_wait_async();
    __syncthreads();
    p ^= 1;
  }

#pragma unroll
  for (int im = 0; im < 2; ++im) {
#pragma unroll
    for (int in = 0; in < 4; ++in) {
      int col = n0 + wn * 64 + in * 16 + lane16;
      float bv = bias[col];
      int rowbase = m0 + wm * 32 + im * 16 + laneHi * 8;
#pragma unroll
      for (int r = 0; r < 8; ++r) {
        out[(long)(rowbase + r) * N + col] = acc[im][in][r] * sc + bv;
      }
    }
  }
  (void)row; (void)coff;
}

extern "C" void kernel_launch(void* const* d_in, const int* in_sizes, int n_in,
                              void* d_out, int out_size, void* d_ws, size_t ws_size,
                              hipStream_t stream) {
  const float* x = (const float*)d_in[0];
  const float* w = (const float*)d_in[1];
  const float* bias = (const float*)d_in[2];
  float* out = (float*)d_out;

  const int N = in_sizes[2];
  const int K = in_sizes[1] / N;
  const int M = in_sizes[0] / K;

  unsigned int* slots = (unsigned int*)d_ws;
  unsigned char* xq = (unsigned char*)d_ws + WS_XQ_OFF;
  unsigned char* wqd = xq + (long)M * K;

  ll_init_kernel<<<1, 32, 0, stream>>>(slots);
  long nx4 = ((long)M * K) / 4, nw4 = ((long)N * K) / 4;
  ll_amax_kernel<<<2048, 256, 0, stream>>>(x, nx4, slots + 0);
  ll_amax_kernel<<<2048, 256, 0, stream>>>(w, nw4, slots + 1);

  ll_quant_kernel<<<4096, 256, 0, stream>>>(x, nx4, slots + 0, (unsigned int*)xq);
  ll_quant_kernel<<<4096, 256, 0, stream>>>(w, nw4, slots + 1, (unsigned int*)wqd);

  dim3 grid(N / BN, M / BM);
  ll_gemm_fp8_kernel<<<grid, 256, 0, stream>>>(xq, wqd, bias, slots, out, M, N, K);
}
